// GraphMLP_26268019982829
// MI455X (gfx1250) — compile-verified, matrix operand order checked by reading, not run on hardware
//
#include <hip/hip_runtime.h>
#include <hip/hip_bf16.h>
#include <stdint.h>

// ---------------------------------------------------------------------------
// GraphMLP pipeline for MI455X (gfx1250), wave32 + V_WMMA_F32_16X16X32_BF16.
//
//  * segment_mean is linear -> dense normalized adjacency A (650x650, exact
//    integer-count float atomics => deterministic), then mean = A @ x.
//  * pairs @ Wp1.T factors through the repeat/tile structure:
//    aproj = attr @ Wp1[:, :512].T (+bp1), oproj = obj @ Wp1[:, 512:].T.
//  * Fused pair kernel, 64 pair rows per block:
//    phase 2 uses 4(m) x 2(n) register blocking: each Wp2 fragment feeds 4
//    WMMAs (4x L2 reuse -> ~3 GB total Wp2 traffic) and each v fragment from
//    LDS feeds 2 WMMAs (halved LDS traffic vs 1-n blocking).
// ---------------------------------------------------------------------------

#define NN    650
#define NATT  250
#define NOBJ  400
#define EE    100000
#define DIN   512
#define DHID  4096
#define FEAT  2048
#define EMB   800
#define BB    256
#define KADJ  672        // 650 padded to multiple of 32
#define P1    1200
#define P1P   1216       // 1200 padded to multiple of 32
#define NPAIR 100000
#define RT    64         // pair rows per fused block (4 m-tiles)

typedef __attribute__((ext_vector_type(16))) __bf16 bf16x16;
typedef __attribute__((ext_vector_type(8)))  float  f32x8;

__device__ __forceinline__ __bf16 f2bf(float f) {
  union { float f; uint32_t u; } in; in.f = f;
  uint32_t r = in.u + 0x7FFFu + ((in.u >> 16) & 1u);
  union { uint16_t s; __bf16 b; } out; out.s = (uint16_t)(r >> 16);
  return out.b;
}

// One 16x32 bf16 operand fragment (per-lane 16 values, two b128 loads).
// rowp = this lane's row base (k-contiguous, row stride multiple of 8).
// K slots per ISA 16-bit A-matrix table:
//   lane<16 : k = k0 + {0..7, 16..23};  lane>=16 : k = k0 + {8..15, 24..31}
__device__ __forceinline__ bf16x16 load_frag(const __bf16* rowp, int k0, int half) {
  union { uint4 u[2]; bf16x16 v; } c;
  c.u[0] = *(const uint4*)(rowp + k0 + 8 * half);
  c.u[1] = *(const uint4*)(rowp + k0 + 16 + 8 * half);
  return c.v;
}

__device__ __forceinline__ f32x8 wmma_bf16(bf16x16 a, bf16x16 b, f32x8 c) {
  return __builtin_amdgcn_wmma_f32_16x16x32_bf16(false, a, false, b, (short)0, c,
                                                 false, false);
}

// ---------------------------------------------------------------------------
// Utility kernels
// ---------------------------------------------------------------------------
__global__ void k_zero(float4* p, long n) {
  long i = (long)blockIdx.x * blockDim.x + threadIdx.x;
  long stride = (long)gridDim.x * blockDim.x;
  float4 z = {0.f, 0.f, 0.f, 0.f};
  for (; i < n; i += stride) p[i] = z;
}

__global__ void k_adj(const int* __restrict__ src, const int* __restrict__ dst,
                      float* __restrict__ A, float* __restrict__ deg) {
  int e = blockIdx.x * blockDim.x + threadIdx.x;
  if (e >= EE) return;
  atomicAdd(&A[(size_t)dst[e] * KADJ + src[e]], 1.0f);  // exact integer counts
  atomicAdd(&deg[dst[e]], 1.0f);
}

__global__ void k_adj_norm(const float* __restrict__ A, const float* __restrict__ deg,
                           __bf16* __restrict__ Abf) {
  int i = blockIdx.x * blockDim.x + threadIdx.x;
  if (i >= NN * KADJ) return;
  float d = deg[i / KADJ];
  d = d > 1.f ? d : 1.f;
  Abf[i] = f2bf(A[i] / d);
}

// dst[c*ldd + r] = bf16(src[r*cols + c])   (f32 row-major -> bf16 transposed)
__global__ void k_transpose_bf(const float* __restrict__ src, __bf16* __restrict__ dst,
                               int rows, int cols, int ldd) {
  long total = (long)rows * cols;
  long stride = (long)gridDim.x * blockDim.x;
  for (long i = (long)blockIdx.x * blockDim.x + threadIdx.x; i < total; i += stride) {
    int r = (int)(i / cols), c = (int)(i - (long)r * cols);
    dst[(size_t)c * ldd + r] = f2bf(src[i]);
  }
}

__global__ void k_convert_bf(const float* __restrict__ src, int lds,
                             __bf16* __restrict__ dst, int ldd, int rows, int cols) {
  long total = (long)rows * cols;
  long stride = (long)gridDim.x * blockDim.x;
  for (long i = (long)blockIdx.x * blockDim.x + threadIdx.x; i < total; i += stride) {
    int r = (int)(i / cols), c = (int)(i - (long)r * cols);
    dst[(size_t)r * ldd + c] = f2bf(src[(size_t)r * lds + c]);
  }
}

// dst[r][0:ca]=a, dst[r][ca:ca+cb]=b (f32 -> bf16), for [mean|x] and [Wl|Wr]
__global__ void k_concat_bf(const float* __restrict__ a, int lda, int ca,
                            const float* __restrict__ b, int ldb, int cb,
                            __bf16* __restrict__ dst, int ldd, int rows) {
  int cc = ca + cb;
  long total = (long)rows * cc;
  long stride = (long)gridDim.x * blockDim.x;
  for (long i = (long)blockIdx.x * blockDim.x + threadIdx.x; i < total; i += stride) {
    int r = (int)(i / cc), c = (int)(i - (long)r * cc);
    float v = (c < ca) ? a[(size_t)r * lda + c] : b[(size_t)r * ldb + (c - ca)];
    dst[(size_t)r * ldd + c] = f2bf(v);
  }
}

// ---------------------------------------------------------------------------
// Generic GEMM: C[M,N] = act(A[M,K] @ B[N,K]^T + bias), bf16 in, f32 acc.
// One wave per 16x16 tile; operands are L2-resident. K multiple of 32.
// Buffers are padded so tile-rounded reads stay in-bounds.
// ---------------------------------------------------------------------------
template <bool RELU, bool OUTBF>
__global__ __launch_bounds__(256)
void gemm_bt_kernel(const __bf16* __restrict__ A, int lda,
                    const __bf16* __restrict__ B, int ldb,
                    const float* __restrict__ bias,
                    void* __restrict__ Cv, int ldc,
                    int Mreal, int Nreal, int K, int ntN, int tiles) {
  int wave = blockIdx.x * (blockDim.x >> 5) + (threadIdx.x >> 5);
  if (wave >= tiles) return;
  int lane = threadIdx.x & 31;
  int half = lane >> 4, lrow = lane & 15;
  int mt = wave / ntN, nt = wave - mt * ntN;

  const __bf16* arow = A + (size_t)(mt * 16 + lrow) * lda;
  const __bf16* brow = B + (size_t)(nt * 16 + lrow) * ldb;
  f32x8 acc = {0.f, 0.f, 0.f, 0.f, 0.f, 0.f, 0.f, 0.f};
  for (int k0 = 0; k0 < K; k0 += 32)
    acc = wmma_bf16(load_frag(arow, k0, half), load_frag(brow, k0, half), acc);

  int n = nt * 16 + lrow;                // this lane's output column
  if (n >= Nreal) return;                // after WMMA: EXEC full during matrix op
  float bv = bias ? bias[n] : 0.f;
#pragma unroll
  for (int r = 0; r < 8; ++r) {          // VGPR r -> row m = r + 8*half
    int m = mt * 16 + r + 8 * half;
    if (m >= Mreal) continue;
    float v = acc[r] + bv;
    if (RELU) v = v > 0.f ? v : 0.f;
    if (OUTBF) ((__bf16*)Cv)[(size_t)m * ldc + n] = f2bf(v);
    else       ((float*)Cv)[(size_t)m * ldc + n] = v;
  }
}

// ---------------------------------------------------------------------------
// Standalone LayerNorm (image branch): x[256,800] -> bf16 out with g,b
// ---------------------------------------------------------------------------
__global__ __launch_bounds__(256)
void k_ln_bf(const float* __restrict__ x, const float* __restrict__ g,
             const float* __restrict__ b, __bf16* __restrict__ out) {
  int row = blockIdx.x, tid = threadIdx.x;
  __shared__ float red[2][8];
  float s = 0.f, s2 = 0.f;
  for (int c = tid; c < EMB; c += 256) {
    float v = x[(size_t)row * EMB + c];
    s += v; s2 += v * v;
  }
  for (int off = 16; off; off >>= 1) { s += __shfl_xor(s, off); s2 += __shfl_xor(s2, off); }
  int wave = tid >> 5, lane = tid & 31;
  if (lane == 0) { red[0][wave] = s; red[1][wave] = s2; }
  __syncthreads();
  if (wave == 0) {
    s  = lane < 8 ? red[0][lane] : 0.f;
    s2 = lane < 8 ? red[1][lane] : 0.f;
    for (int off = 4; off; off >>= 1) { s += __shfl_xor(s, off); s2 += __shfl_xor(s2, off); }
    if (lane == 0) { red[0][0] = s; red[1][0] = s2; }
  }
  __syncthreads();
  float mean = red[0][0] * (1.f / EMB);
  float var  = red[1][0] * (1.f / EMB) - mean * mean;
  float rstd = rsqrtf(var + 1e-5f);
  for (int c = tid; c < EMB; c += 256)
    out[(size_t)row * EMB + c] =
        f2bf((x[(size_t)row * EMB + c] - mean) * rstd * g[c] + b[c]);
}

// ---------------------------------------------------------------------------
// Fused pair kernel, RT=64 pair rows per block, 512 threads (16 waves):
//  1) v[64,1216] = relu(aproj[a(r)] + oproj[o(r)])          (LDS bf16, 156KB)
//  2) q[64,800]  = v @ Wp2^T + bp2 : 4(m) x 2(n) register blocking per wave;
//     B frag (Wp2, L2) reused 4x, A frag (v, LDS) reused 2x -> bf16 LDS
//  3) in-place LayerNorm of q rows (f32 stats, bf16 store)
//  4) out[:, r0..r0+63] = i_ln @ q_ln^T : per wave, one A frag (i_ln) feeds
//     4 n-tile WMMAs; 16 waves cover the 16 m-tiles of the 256 image rows.
// LDS total = 64*800*2 + 64*1216*2 = 258048 B  (< 320KB/WGP -> 1 block/WGP)
// ---------------------------------------------------------------------------
#define SM_Q  (RT * EMB * 2)
#define SM_V  (RT * P1P * 2)
#define SM_TOTAL (SM_Q + SM_V)

__global__ __launch_bounds__(512)
void pair_fused_kernel(const float* __restrict__ aproj,   // 256 x 1216 (has bp1)
                       const float* __restrict__ oproj,   // 400 x 1216
                       const __bf16* __restrict__ Wp2,    // 800 x 1216 bf16
                       const float* __restrict__ bp2,
                       const float* __restrict__ gp,
                       const float* __restrict__ bpn,
                       const __bf16* __restrict__ iln,    // 256 x 800 bf16
                       float* __restrict__ out) {         // 256 x 100000
  extern __shared__ char smem[];
  __bf16* q_s = (__bf16*)smem;            // RT x 800  (pre- and post-LN)
  __bf16* v_s = (__bf16*)(smem + SM_Q);   // RT x 1216

  int r0 = blockIdx.x * RT;
  int tid = threadIdx.x;

  // phase 1: v = relu(aproj[a] + oproj[o]) ; pad cols / tail rows -> 0
  for (int idx = tid; idx < RT * P1P; idx += 512) {
    int n = idx / P1P, c = idx - n * P1P;
    int rr = r0 + n;
    float v = 0.f;
    if (c < P1 && rr < NPAIR) {
      int a = rr / NOBJ, o = rr - a * NOBJ;
      v = aproj[(size_t)a * P1P + c] + oproj[(size_t)o * P1P + c];
      v = v > 0.f ? v : 0.f;
    }
    v_s[idx] = f2bf(v);
  }
  __syncthreads();

  int wave = tid >> 5, lane = tid & 31, half = lane >> 4, lrow = lane & 15;

  // phase 2: q = v @ Wp2^T + bp2  (M=64, N=800). 50 n-tiles = 25 pairs
  // (nt, nt+25) spread over 16 waves; 4x2 register blocking.
  for (int base = wave; base < 25; base += 16) {
    const __bf16* brow0 = Wp2 + (size_t)(base * 16 + lrow) * P1P;
    const __bf16* brow1 = Wp2 + (size_t)((base + 25) * 16 + lrow) * P1P;
    f32x8 acc[2][4];
#pragma unroll
    for (int j = 0; j < 2; ++j)
#pragma unroll
      for (int mt = 0; mt < 4; ++mt)
        acc[j][mt] = f32x8{0.f,0.f,0.f,0.f,0.f,0.f,0.f,0.f};
    for (int k0 = 0; k0 < P1P; k0 += 32) {
      bf16x16 bf0 = load_frag(brow0, k0, half);       // 2 B frags from L2 ...
      bf16x16 bf1 = load_frag(brow1, k0, half);
#pragma unroll
      for (int mt = 0; mt < 4; ++mt) {                // ... 4 A frags from LDS
        bf16x16 af = load_frag(v_s + (mt * 16 + lrow) * P1P, k0, half);
        acc[0][mt] = wmma_bf16(af, bf0, acc[0][mt]);  // each A frag -> 2 WMMAs
        acc[1][mt] = wmma_bf16(af, bf1, acc[1][mt]);
      }
    }
#pragma unroll
    for (int j = 0; j < 2; ++j) {
      int n = (base + 25 * j) * 16 + lrow;
      float bv = bp2[n];
#pragma unroll
      for (int mt = 0; mt < 4; ++mt)
#pragma unroll
        for (int r = 0; r < 8; ++r)
          q_s[(mt * 16 + r + 8 * half) * EMB + n] = f2bf(acc[j][mt][r] + bv);
    }
  }
  __syncthreads();

  // phase 3: in-place LayerNorm of the 64 q rows (4 rows per wave)
  for (int m = wave * 4; m < wave * 4 + 4; ++m) {
    float s = 0.f, s2 = 0.f;
    for (int c = lane; c < EMB; c += 32) {
      float x = (float)q_s[m * EMB + c];
      s += x; s2 += x * x;
    }
    for (int off = 16; off; off >>= 1) { s += __shfl_xor(s, off); s2 += __shfl_xor(s2, off); }
    float mean = s * (1.f / EMB);
    float var  = s2 * (1.f / EMB) - mean * mean;
    float rstd = rsqrtf(var + 1e-5f);
    for (int c = lane; c < EMB; c += 32) {
      float x = (float)q_s[m * EMB + c];
      q_s[m * EMB + c] = f2bf((x - mean) * rstd * gp[c] + bpn[c]);
    }
  }
  __syncthreads();

  // phase 4: out tile = i_ln @ q_ln^T  (M=256 -> wave=mt, N=64 -> 4 n-tiles)
  {
    int mt = wave;                                     // 16 waves = 16 m-tiles
    const __bf16* arow = iln + (size_t)(mt * 16 + lrow) * EMB;
    f32x8 acc[4];
#pragma unroll
    for (int nt = 0; nt < 4; ++nt) acc[nt] = f32x8{0.f,0.f,0.f,0.f,0.f,0.f,0.f,0.f};
    for (int k0 = 0; k0 < EMB; k0 += 32) {
      bf16x16 afrag = load_frag(arow, k0, half);      // one i_ln fetch ...
#pragma unroll
      for (int nt = 0; nt < 4; ++nt) {                // ... 4 WMMAs of reuse
        bf16x16 bfrag = load_frag(q_s + (nt * 16 + lrow) * EMB, k0, half);
        acc[nt] = wmma_bf16(afrag, bfrag, acc[nt]);
      }
    }
#pragma unroll
    for (int nt = 0; nt < 4; ++nt) {
      int col = r0 + nt * 16 + lrow;
      if (col < NPAIR) {
#pragma unroll
        for (int r = 0; r < 8; ++r) {
          int mrow = mt * 16 + r + 8 * half;
          out[(size_t)mrow * NPAIR + col] = acc[nt][r];
        }
      }
    }
  }
}

// ---------------------------------------------------------------------------
// Host orchestration
// ---------------------------------------------------------------------------
extern "C" void kernel_launch(void* const* d_in, const int* in_sizes, int n_in,
                              void* d_out, int out_size, void* d_ws, size_t ws_size,
                              hipStream_t stream) {
  const float* img  = (const float*)d_in[0];
  const float* nodes= (const float*)d_in[1];
  const int*   esrc = (const int*)d_in[2];
  const int*   edst = (const int*)d_in[3];
  const float* W1l  = (const float*)d_in[4];
  const float* b1   = (const float*)d_in[5];
  const float* W1r  = (const float*)d_in[6];
  const float* W2l  = (const float*)d_in[7];
  const float* b2   = (const float*)d_in[8];
  const float* W2r  = (const float*)d_in[9];
  const float* Wp1  = (const float*)d_in[10];
  const float* bp1  = (const float*)d_in[11];
  const float* Wp2f = (const float*)d_in[12];
  const float* bp2  = (const float*)d_in[13];
  const float* gp   = (const float*)d_in[14];
  const float* bpn  = (const float*)d_in[15];
  const float* Wi1  = (const float*)d_in[16];
  const float* bi1  = (const float*)d_in[17];
  const float* Wi2  = (const float*)d_in[18];
  const float* bi2  = (const float*)d_in[19];
  const float* Wi3  = (const float*)d_in[20];
  const float* bi3  = (const float*)d_in[21];
  const float* gi   = (const float*)d_in[22];
  const float* binn = (const float*)d_in[23];
  (void)in_sizes; (void)n_in; (void)out_size; (void)ws_size;

  char* ws = (char*)d_ws;
  size_t off = 0;
  auto alloc = [&](size_t bytes) -> char* {
    char* p = ws + off;
    off = (off + bytes + 255) & ~(size_t)255;
    return p;
  };
  // padded-row buffers so tile-rounded WMMA reads stay in bounds; ws is zeroed.
  float*  A_f     = (float*) alloc((size_t)656 * KADJ * 4);
  float*  deg     = (float*) alloc(656 * 4);
  __bf16* A_bf    = (__bf16*)alloc((size_t)656 * KADJ * 2);
  __bf16* nodesT  = (__bf16*)alloc((size_t)512 * KADJ * 2);
  float*  mean_x  = (float*) alloc((size_t)656 * DIN * 4);
  __bf16* X1_bf   = (__bf16*)alloc((size_t)656 * 1024 * 2);
  __bf16* Wcat    = (__bf16*)alloc((size_t)DHID * 1024 * 2);   // reused: [W1l|W1r] then [W2l|W2r]
  float*  h_f     = (float*) alloc((size_t)656 * DHID * 4);
  __bf16* hT_bf   = (__bf16*)alloc((size_t)DHID * KADJ * 2);
  float*  mean_h  = (float*) alloc((size_t)656 * DHID * 4);
  __bf16* X2_bf   = (__bf16*)alloc((size_t)656 * 8192 * 2);
  float*  on2_f   = (float*) alloc((size_t)656 * DIN * 4);
  __bf16* on2_bf  = (__bf16*)alloc((size_t)656 * DIN * 2);
  __bf16* Wp1_bf  = (__bf16*)alloc((size_t)P1P * 1024 * 2);
  float*  aproj   = (float*) alloc((size_t)256 * P1P * 4);
  float*  oproj   = (float*) alloc((size_t)NOBJ * P1P * 4);
  __bf16* Wp2_bf  = (__bf16*)alloc((size_t)EMB * P1P * 2);
  __bf16* img_bf  = (__bf16*)alloc((size_t)BB * FEAT * 2);
  __bf16* Wi1_bf  = (__bf16*)alloc((size_t)768 * FEAT * 2);
  __bf16* i1_bf   = (__bf16*)alloc((size_t)BB * 768 * 2);
  __bf16* Wi2_bf  = (__bf16*)alloc((size_t)1008 * 768 * 2);
  __bf16* i2_bf   = (__bf16*)alloc((size_t)BB * 1024 * 2);
  __bf16* Wi3_bf  = (__bf16*)alloc((size_t)EMB * 1024 * 2);
  float*  i3_f    = (float*) alloc((size_t)BB * EMB * 4);
  __bf16* iln_bf  = (__bf16*)alloc((size_t)BB * EMB * 2);
  size_t ws_used = off;

  auto gemm = [&](const __bf16* A, int lda, const __bf16* B, int ldb,
                  const float* bias, void* C, int ldc,
                  int M, int N, int K, bool relu, bool outbf) {
    int mt = (M + 15) / 16, nt = (N + 15) / 16;
    int tiles = mt * nt;
    int blocks = (tiles + 7) / 8;
    if (relu && outbf)
      gemm_bt_kernel<true, true><<<blocks, 256, 0, stream>>>(A, lda, B, ldb, bias, C, ldc, M, N, K, nt, tiles);
    else if (relu)
      gemm_bt_kernel<true, false><<<blocks, 256, 0, stream>>>(A, lda, B, ldb, bias, C, ldc, M, N, K, nt, tiles);
    else if (outbf)
      gemm_bt_kernel<false, true><<<blocks, 256, 0, stream>>>(A, lda, B, ldb, bias, C, ldc, M, N, K, nt, tiles);
    else
      gemm_bt_kernel<false, false><<<blocks, 256, 0, stream>>>(A, lda, B, ldb, bias, C, ldc, M, N, K, nt, tiles);
  };

  const int EW = 2048;  // grid for elementwise grid-stride kernels

  // 0) zero all scratch (guarantees zero pads everywhere)
  k_zero<<<4096, 256, 0, stream>>>((float4*)ws, (long)(ws_used / 16));
  // 1) adjacency: counts + degrees (exact -> deterministic), normalize -> bf16
  k_adj<<<(EE + 255) / 256, 256, 0, stream>>>(esrc, edst, A_f, deg);
  k_adj_norm<<<(NN * KADJ + 255) / 256, 256, 0, stream>>>(A_f, deg, A_bf);
  // 2) nodes^T (bf16, K padded to 672)
  k_transpose_bf<<<EW, 256, 0, stream>>>(nodes, nodesT, NN, DIN, KADJ);
  // 3) mean_x = A @ nodes        (650x672x512)
  gemm(A_bf, KADJ, nodesT, KADJ, nullptr, mean_x, DIN, NN, DIN, KADJ, false, false);
  // 4) h = relu([mean_x|nodes] @ [W1l|W1r]^T + b1)   (650x1024x4096)
  k_concat_bf<<<EW, 256, 0, stream>>>(mean_x, DIN, DIN, nodes, DIN, DIN, X1_bf, 1024, NN);
  k_concat_bf<<<EW, 256, 0, stream>>>(W1l, DIN, DIN, W1r, DIN, DIN, Wcat, 1024, DHID);
  gemm(X1_bf, 1024, Wcat, 1024, b1, h_f, DHID, NN, DHID, 1024, true, false);
  // 5) mean_h = A @ h            (650x672x4096)
  k_transpose_bf<<<EW, 256, 0, stream>>>(h_f, hT_bf, NN, DHID, KADJ);
  k_convert_bf<<<EW, 256, 0, stream>>>(h_f, DHID, X2_bf + DHID, 8192, NN, DHID);
  gemm(A_bf, KADJ, hT_bf, KADJ, nullptr, mean_h, DHID, NN, DHID, KADJ, false, false);
  k_convert_bf<<<EW, 256, 0, stream>>>(mean_h, DHID, X2_bf, 8192, NN, DHID);
  // 6) out_nodes = [mean_h|h] @ [W2l|W2r]^T + b2    (650x8192x512)
  k_concat_bf<<<EW, 256, 0, stream>>>(W2l, DHID, DHID, W2r, DHID, DHID, Wcat, 8192, DIN);
  gemm(X2_bf, 8192, Wcat, 8192, b2, on2_f, DIN, NN, DIN, 8192, false, false);
  k_convert_bf<<<EW, 256, 0, stream>>>(on2_f, DIN, on2_bf, DIN, NN, DIN);
  // 7) pair projections (Wp1 split at col 512); bp1 folded into aproj
  k_convert_bf<<<EW, 256, 0, stream>>>(Wp1, 1024, Wp1_bf, 1024, P1, 1024);
  gemm(on2_bf, DIN, Wp1_bf, 1024, bp1, aproj, P1P, NATT, P1, DIN, false, false);
  gemm(on2_bf + (size_t)NATT * DIN, DIN, Wp1_bf + DIN, 1024, nullptr, oproj, P1P, NOBJ, P1, DIN, false, false);
  k_convert_bf<<<EW, 256, 0, stream>>>(Wp2f, P1, Wp2_bf, P1P, EMB, P1);
  // 8) image branch
  k_convert_bf<<<EW, 256, 0, stream>>>(img, FEAT, img_bf, FEAT, BB, FEAT);
  k_convert_bf<<<EW, 256, 0, stream>>>(Wi1, FEAT, Wi1_bf, FEAT, 768, FEAT);
  gemm(img_bf, FEAT, Wi1_bf, FEAT, bi1, i1_bf, 768, BB, 768, FEAT, true, true);
  k_convert_bf<<<EW, 256, 0, stream>>>(Wi2, 768, Wi2_bf, 768, 1000, 768);
  gemm(i1_bf, 768, Wi2_bf, 768, bi2, i2_bf, 1024, BB, 1000, 768, true, true);
  k_convert_bf<<<EW, 256, 0, stream>>>(Wi3, 1000, Wi3_bf, 1024, EMB, 1000);
  gemm(i2_bf, 1024, Wi3_bf, 1024, bi3, i3_f, EMB, BB, EMB, 1024, false, false);
  k_ln_bf<<<BB, 256, 0, stream>>>(i3_f, gi, binn, iln_bf);
  // 9) fused pair MLP + LN + outer product (the 233 GFLOP stage)
  pair_fused_kernel<<<(NPAIR + RT - 1) / RT, 512, SM_TOTAL, stream>>>(
      aproj, oproj, Wp2_bf, bp2, gp, bpn, iln_bf, (float*)d_out);
}
